// LinAngularAttention_59365037965653
// MI455X (gfx1250) — matrix operand order repaired and checked by reading, not compiled, not run on hardware
//
#include <hip/hip_runtime.h>

#define DEV __device__ __forceinline__

typedef __attribute__((ext_vector_type(16))) __bf16 v16bf;
typedef __attribute__((ext_vector_type(8)))  float  v8f;
typedef __attribute__((ext_vector_type(8)))  unsigned int v8u;

namespace laa {

constexpr int Nb = 8, Lb = 8192, Cb = 256, Hh = 4, Dd = 64;
constexpr float INV_PI = 0.31830988618379067f;

DEV unsigned short f2bf(float f) {
    unsigned int u = __builtin_bit_cast(unsigned int, f);
    u += 0x7FFFu + ((u >> 16) & 1u);          // round-to-nearest-even
    return (unsigned short)(u >> 16);
}

// A fragment: M x K tile, A row-major (K contiguous), 16-bit elements.
// ISA layout: lane m = lane&15, half = lane>>4; VGPR j holds K pair
// base = (j<4?0:16) + half*8 + 2*(j&3).  (compiler fuses into 2x b128 loads)
DEV v16bf frag_a(const unsigned short* A, int lda, int row0, int k0, int lane) {
    const unsigned int* A32 = (const unsigned int*)A;
    const int m = lane & 15, hf = lane >> 4, lda2 = lda >> 1;
    v8u u;
#pragma unroll
    for (int j = 0; j < 8; ++j) {
        const int kb = ((j < 4) ? 0 : 16) + hf * 8 + 2 * (j & 3);
        u[j] = A32[(size_t)(row0 + m) * lda2 + ((k0 + kb) >> 1)];
    }
    return __builtin_bit_cast(v16bf, u);
}

// B fragment from B^T stored row-major as (N x K), 16-bit elements.
// ISA layout: lane n = lane&15, khalf = lane>>4; VGPR j holds K = (j<4?0:16) + khalf*8 + 2*(j&3) (+1).
DEV v16bf frag_bt(const unsigned short* BT, int ldb, int col0, int k0, int lane) {
    const unsigned int* B32 = (const unsigned int*)BT;
    const int n = lane & 15, kh = lane >> 4, ldb2 = ldb >> 1;
    v8u u;
#pragma unroll
    for (int j = 0; j < 8; ++j) {
        const int kb = ((j < 4) ? 0 : 16) + kh * 8 + 2 * (j & 3);
        u[j] = B32[(size_t)(col0 + n) * ldb2 + ((k0 + kb) >> 1)];
    }
    return __builtin_bit_cast(v16bf, u);
}

DEV v8f wmma_bf(v16bf a, v16bf b, v8f c) {
    return __builtin_amdgcn_wmma_f32_16x16x32_bf16(false, a, false, b, (short)0, c, false, false);
}

} // namespace laa

// ---------------- fp32 -> bf16 conversion ----------------
__global__ __launch_bounds__(256) void k_cvt_bf16(const float* __restrict__ src,
                                                  unsigned short* __restrict__ dst, int n) {
    const int i = blockIdx.x * blockDim.x + threadIdx.x;
    if (i < n) dst[i] = laa::f2bf(src[i]);
}

// ---------------- QKV GEMM: (65536x256) x (256x768) ----------------
// Each wave computes a 16x64 strip: A fragment reused across 4 N-tiles.
// Output scattered into q_raw / k_raw / v in (N,H,L,D) layout (fp32).
__global__ __launch_bounds__(256) void k_qkv_gemm(
    const unsigned short* __restrict__ xbf,    // 65536 x 256 bf16, row-major
    const unsigned short* __restrict__ wbf,    // 768 x 256 bf16  (= B^T row-major)
    float* __restrict__ qraw, float* __restrict__ kraw, float* __restrict__ vbuf) {
    using namespace laa;
    const int lane = threadIdx.x & 31;
    const int wave = threadIdx.x >> 5;
    const int row0 = (blockIdx.x * 8 + wave) * 16;
    const int col0 = blockIdx.y * 64;          // 64-wide strip (one s,h group)
    v8f acc[4] = {};
#pragma unroll
    for (int kk = 0; kk < Cb; kk += 32) {
        v16bf a = frag_a(xbf, Cb, row0, kk, lane);
#pragma unroll
        for (int t = 0; t < 4; ++t) {
            v16bf b = frag_bt(wbf, Cb, col0 + t * 16, kk, lane);
            acc[t] = wmma_bf(a, b, acc[t]);
        }
    }
    const int s = col0 >> 8, h = (col0 >> 6) & 3;
    float* dst = (s == 0) ? qraw : (s == 1) ? kraw : vbuf;
#pragma unroll
    for (int t = 0; t < 4; ++t) {
        const int d = t * 16 + (lane & 15);
#pragma unroll
        for (int j = 0; j < 8; ++j) {
            const int m = row0 + j + 8 * (lane >> 4);
            const int n = m >> 13, l = m & (Lb - 1);
            dst[(((size_t)n * Hh + h) * Lb + l) * Dd + d] = acc[t][j];
        }
    }
}

// ---------------- mask + l2norm + pack, LDS-transposed coalesced kT/vT ----------------
// One block: one (n,h), 64 rows of L, full D=64. Phase 1 normalizes rows and stages
// bf16 into padded LDS tiles; phase 2 streams coalesced b32 writes along L.
__global__ __launch_bounds__(256) void k_norm_pack(
    const float* __restrict__ qraw, const float* __restrict__ kraw,
    const float* __restrict__ vbuf, const int* __restrict__ amask,
    unsigned short* __restrict__ qbf,   // (NH, L, D) row-major
    unsigned short* __restrict__ kT,    // (NH, D, L)
    unsigned short* __restrict__ vT) {  // (NH, D, L)
    using namespace laa;
    __shared__ unsigned short sK[64][66];   // pad 66 -> stride 33 words, conflict-free
    __shared__ unsigned short sV[64][66];
    const int tid = threadIdx.x;
    const int lane = tid & 31, wave = tid >> 5;     // 8 waves
    const int nh = blockIdx.x >> 7;                 // 32 nh x 128 l-tiles
    const int l0 = (blockIdx.x & 127) * 64;
    const int n = nh >> 2;
    const size_t rb = ((size_t)nh * Lb + l0) * Dd;
#pragma unroll
    for (int r = 0; r < 8; ++r) {
        const int ll = wave + 8 * r;
        const int l = l0 + ll;
        const size_t base = rb + (size_t)ll * Dd;
        const int d0 = lane, d1 = lane + 32;
        float q0 = qraw[base + d0], q1 = qraw[base + d1];
        if (amask[n * Lb + l] == 0) { q0 = -1e9f; q1 = -1e9f; }
        const float k0 = kraw[base + d0], k1 = kraw[base + d1];
        const float v0 = vbuf[base + d0], v1 = vbuf[base + d1];
        float sq = q0 * q0 + q1 * q1;
        float sk = k0 * k0 + k1 * k1;
#pragma unroll
        for (int off = 1; off < 32; off <<= 1) {
            sq += __shfl_xor(sq, off, 32);
            sk += __shfl_xor(sk, off, 32);
        }
        const float rq = rsqrtf(sq), rk = rsqrtf(sk);
        qbf[base + d0] = f2bf(q0 * rq);
        qbf[base + d1] = f2bf(q1 * rq);
        sK[d0][ll] = f2bf(k0 * rk);
        sK[d1][ll] = f2bf(k1 * rk);
        sV[d0][ll] = f2bf(v0);
        sV[d1][ll] = f2bf(v1);
    }
    __syncthreads();
    unsigned int* kT32 = (unsigned int*)kT;
    unsigned int* vT32 = (unsigned int*)vT;
    const int Lb2 = Lb >> 1;
#pragma unroll
    for (int it = 0; it < 8; ++it) {
        const int d = wave + 8 * it;
        const unsigned int kk2 = *(const unsigned int*)&sK[d][2 * lane];
        const unsigned int vv2 = *(const unsigned int*)&sV[d][2 * lane];
        const size_t o = (size_t)(nh * Dd + d) * Lb2 + (l0 >> 1) + lane;
        kT32[o] = kk2;
        vT32[o] = vv2;
    }
}

// ---------------- kv = k^T v  per (n,h): (64x8192)x(8192x64) ----------------
// Output stored transposed (e,d) and pre-scaled by 1/pi, so it is B^T for q@kv.
__global__ __launch_bounds__(512) void k_kv_gemm(
    const unsigned short* __restrict__ kT, const unsigned short* __restrict__ vT,
    unsigned short* __restrict__ kvT) {
    using namespace laa;
    const int lane = threadIdx.x & 31;
    const int wave = threadIdx.x >> 5;     // 0..15 -> 4x4 tile grid of 64x64
    const int nh = blockIdx.x;
    const int tm = wave >> 2, tn = wave & 3;
    const unsigned short* A = kT + (size_t)nh * Dd * Lb;
    const unsigned short* B = vT + (size_t)nh * Dd * Lb;
    v8f acc = {};
    for (int kk = 0; kk < Lb; kk += 32) {
        v16bf a = frag_a(A, Lb, tm * 16, kk, lane);
        v16bf b = frag_bt(B, Lb, tn * 16, kk, lane);
        acc = wmma_bf(a, b, acc);
    }
    const int e = tn * 16 + (lane & 15);
#pragma unroll
    for (int j = 0; j < 8; ++j) {
        const int d = tm * 16 + j + 8 * (lane >> 4);
        kvT[(size_t)nh * Dd * Dd + (size_t)e * Dd + d] = f2bf(acc[j] * INV_PI);
    }
}

// ---------------- out = l2norm(0.5*v + q@kv/pi) + dconv(v), pack bf16 ----------------
// Depthwise 9-tap conv fused into epilogue (neighbor rows are cache-resident).
__global__ __launch_bounds__(256) void k_attn_out(
    const unsigned short* __restrict__ qbf, const unsigned short* __restrict__ kvT,
    const float* __restrict__ vbuf, const float* __restrict__ w9,
    unsigned short* __restrict__ attn) {   // (N*L, 256) bf16, row-major
    using namespace laa;
    const int lane = threadIdx.x & 31;
    const int wave = threadIdx.x >> 5;
    const int nh = blockIdx.x >> 6;
    const int l0 = ((blockIdx.x & 63) * 8 + wave) * 16;
    const int n = nh >> 2, h = nh & 3;
    const unsigned short* A = qbf + (size_t)nh * Lb * Dd;
    const unsigned short* B = kvT + (size_t)nh * Dd * Dd;
    const size_t vb = (size_t)nh * Lb * Dd;
    float wt[9];
#pragma unroll
    for (int tt = 0; tt < 9; ++tt) wt[tt] = w9[h * 9 + tt];

    v8f acc[4] = {};
#pragma unroll
    for (int kk = 0; kk < Dd; kk += 32) {
        v16bf a = frag_a(A, Dd, l0, kk, lane);       // reused across 4 N-tiles
#pragma unroll
        for (int t = 0; t < 4; ++t) {
            v16bf b = frag_bt(B, Dd, t * 16, kk, lane);
            acc[t] = wmma_bf(a, b, acc[t]);
        }
    }
    const int e = lane & 15, g = lane >> 4;
#pragma unroll
    for (int j = 0; j < 8; ++j) {
        const int l = l0 + j + 8 * g;
        float vals[4], cvs[4];
        float ss = 0.f;
#pragma unroll
        for (int t = 0; t < 4; ++t) {
            const int col = t * 16 + e;
            float cv = 0.f, vc = 0.f;
#pragma unroll
            for (int tt = 0; tt < 9; ++tt) {
                const int l2 = l + tt - 4;
                if (l2 >= 0 && l2 < Lb) {
                    const float vv = vbuf[vb + (size_t)l2 * Dd + col];
                    cv += wt[tt] * vv;
                    if (tt == 4) vc = vv;     // center tap == v[l][col]
                }
            }
            vals[t] = acc[t][j] + 0.5f * vc;
            cvs[t] = cv;
            ss += vals[t] * vals[t];
        }
#pragma unroll
        for (int off = 1; off < 16; off <<= 1) ss += __shfl_xor(ss, off, 32); // 16-lane groups
        const float rn = rsqrtf(ss);
        const size_t orow = ((size_t)n * Lb + l) * Cb + (size_t)h * Dd;
#pragma unroll
        for (int t = 0; t < 4; ++t) {
            attn[orow + t * 16 + e] = f2bf(vals[t] * rn + cvs[t]);
        }
    }
}

// ---------------- projection GEMM: (65536x256) x (256x256) + bias ----------------
// 16x64 strip per wave, A fragment reused across 4 N-tiles.
__global__ __launch_bounds__(256) void k_proj_gemm(
    const unsigned short* __restrict__ attn, const unsigned short* __restrict__ wbf,
    const float* __restrict__ bias, float* __restrict__ out) {
    using namespace laa;
    const int lane = threadIdx.x & 31;
    const int wave = threadIdx.x >> 5;
    const int row0 = (blockIdx.x * 8 + wave) * 16;
    const int col0 = blockIdx.y * 64;
    v8f acc[4] = {};
#pragma unroll
    for (int kk = 0; kk < Cb; kk += 32) {
        v16bf a = frag_a(attn, Cb, row0, kk, lane);
#pragma unroll
        for (int t = 0; t < 4; ++t) {
            v16bf b = frag_bt(wbf, Cb, col0 + t * 16, kk, lane);
            acc[t] = wmma_bf(a, b, acc[t]);
        }
    }
#pragma unroll
    for (int t = 0; t < 4; ++t) {
        const int col = col0 + t * 16 + (lane & 15);
        const float bb = bias[col];
#pragma unroll
        for (int j = 0; j < 8; ++j) {
            const int m = row0 + j + 8 * (lane >> 4);
            out[(size_t)m * Cb + col] = acc[t][j] + bb;
        }
    }
}

extern "C" void kernel_launch(void* const* d_in, const int* in_sizes, int n_in,
                              void* d_out, int out_size, void* d_ws, size_t ws_size,
                              hipStream_t stream) {
    using namespace laa;
    const float* x       = (const float*)d_in[0];
    const int*   amask   = (const int*)d_in[1];
    const float* qkv_w   = (const float*)d_in[2];
    const float* proj_w  = (const float*)d_in[3];
    const float* proj_b  = (const float*)d_in[4];
    const float* dconv_w = (const float*)d_in[5];

    constexpr size_t NLC = (size_t)Nb * Lb * Cb;   // 16,777,216 elements

    char* ws = (char*)d_ws;
    size_t off = 0;
    auto take = [&](size_t bytes) -> void* {
        void* p = ws + off;
        off += (bytes + 255) & ~(size_t)255;
        return p;
    };
    unsigned short* xbf    = (unsigned short*)take(NLC * 2);
    unsigned short* qkvwb  = (unsigned short*)take((size_t)768 * 256 * 2);
    unsigned short* projwb = (unsigned short*)take((size_t)256 * 256 * 2);
    float* qraw = (float*)take(NLC * 4);     // dead after k_norm_pack
    float* kraw = (float*)take(NLC * 4);     // dead after k_norm_pack -> reused for attn bf16
    float* vbuf = (float*)take(NLC * 4);
    unsigned short* qbf = (unsigned short*)take(NLC * 2);
    unsigned short* kT  = (unsigned short*)take(NLC * 2);
    unsigned short* vT  = (unsigned short*)take(NLC * 2);
    unsigned short* kvT = (unsigned short*)take((size_t)32 * 64 * 64 * 2);
    if (off > ws_size) return;               // workspace too small: bail deterministically

    unsigned short* attn = (unsigned short*)kraw;  // alias (kraw dead by then)

    k_cvt_bf16<<<65536, 256, 0, stream>>>(x, xbf, (int)NLC);
    k_cvt_bf16<<<768, 256, 0, stream>>>(qkv_w, qkvwb, 768 * 256);
    k_cvt_bf16<<<256, 256, 0, stream>>>(proj_w, projwb, 256 * 256);

    k_qkv_gemm<<<dim3(512, 12), 256, 0, stream>>>(xbf, qkvwb, qraw, kraw, vbuf);
    k_norm_pack<<<4096, 256, 0, stream>>>(qraw, kraw, vbuf, amask, qbf, kT, vT);
    k_kv_gemm<<<32, 512, 0, stream>>>(kT, vT, kvT);
    k_attn_out<<<2048, 256, 0, stream>>>(qbf, kvT, vbuf, dconv_w, attn);
    k_proj_gemm<<<dim3(512, 4), 256, 0, stream>>>(attn, projwb, proj_b, (float*)d_out);
}
